// GraphLayer_35510789603863
// MI455X (gfx1250) — compile-verified, matrix operand order checked by reading, not run on hardware
//
#include <hip/hip_runtime.h>
#include <math.h>

// ---------------- problem constants (fixed by the reference) ----------------
constexpr int Bc   = 16;
constexpr int Nc   = 2000;
constexpr int CIN  = 128;
constexpr int H    = 4;
constexpr int CH   = 32;
constexpr int COUT = H * CH;          // 128
constexpr int Ec   = 32000;
constexpr int BN   = Bc * Nc;         // 32000
constexpr int ETOT = Bc * Ec + BN;    // 544000 (batched edges + self loops)
constexpr float NEG_SLOPE = 0.2f;

// packed W fragments: [hi/lo][nt=8][kt=4][lane=32] x 16 bf16 (32B each)
constexpr int WFRAGS = 8 * 4 * 32;    // 1024 fragments per (hi|lo) plane

typedef __attribute__((ext_vector_type(16))) __bf16 v16bf;
typedef __attribute__((ext_vector_type(8)))  float  v8f;

// ---------------- helpers ----------------
__device__ __forceinline__ float lrelu(float v) {
    return v > 0.0f ? v : NEG_SLOPE * v;
}

// float atomic max via int/uint ordering trick (works across sign classes)
__device__ __forceinline__ void atomicMaxFloat(float* addr, float v) {
    if (v >= 0.0f) atomicMax((int*)addr, __float_as_int(v));
    else           atomicMin((unsigned int*)addr, __float_as_uint(v));
}

// decode batched edge id -> (src, dst) node ids with self loops appended
__device__ __forceinline__ void edge_nodes(int eId, const int* __restrict__ ei,
                                           int& s, int& d) {
    if (eId < Bc * Ec) {
        int b = eId / Ec;
        int e = eId - b * Ec;
        s = ei[e]      + b * Nc;
        d = ei[Ec + e] + b * Nc;
    } else {
        s = d = eId - Bc * Ec;
    }
}

// load 16 fp32 (two contiguous float4 pairs at k0 and k0+16), split into
// bf16 hi/lo fragments following the 16-bit WMMA A/B lane layout
__device__ __forceinline__ void cvt_split16(const float* __restrict__ base, int k0,
                                            v16bf& hi, v16bf& lo) {
    #pragma unroll
    for (int part = 0; part < 2; ++part) {
        float4 f0 = *(const float4*)(base + k0 + part * 16);
        float4 f1 = *(const float4*)(base + k0 + part * 16 + 4);
        float vals[8] = {f0.x, f0.y, f0.z, f0.w, f1.x, f1.y, f1.z, f1.w};
        #pragma unroll
        for (int i = 0; i < 8; ++i) {
            float  f = vals[i];
            __bf16 h = (__bf16)f;
            hi[part * 8 + i] = h;
            lo[part * 8 + i] = (__bf16)(f - (float)h);
        }
    }
}

// ---------------- kernel 0: pack W into per-lane WMMA fragments ----------------
// one thread per fragment: t = ((nt*4)+kt)*32 + lane
__global__ void gat_pack_w(const float* __restrict__ W, v16bf* __restrict__ wpack) {
    int t = blockIdx.x * blockDim.x + threadIdx.x;
    if (t >= WFRAGS) return;
    int lane = t & 31;
    int kt   = (t >> 5) & 3;
    int nt   = t >> 7;
    int lidx = lane & 15;
    int koff = (lane >> 4) * 8;
    const float* wrow = W + (size_t)(nt * 16 + lidx) * CIN;
    v16bf hi, lo;
    cvt_split16(wrow, kt * 32 + koff, hi, lo);
    wpack[t]          = hi;
    wpack[WFRAGS + t] = lo;
}

// ---------------- kernel 1: xp = x @ W^T via bf16x3 WMMA ----------------
// grid: 250 blocks x 256 threads (8 waves). wave w of block b owns node rows
// [(b*8+w)*16, +16) and all 128 output channels. Packed W fragments (64KB)
// are staged into LDS once per block with async global->LDS copies.
__global__ __launch_bounds__(256)
void gat_gemm_wmma(const float* __restrict__ x,
                   const v16bf* __restrict__ wpack,
                   float* __restrict__ xp) {
    __shared__ v16bf sW[2 * WFRAGS];   // 64 KB: hi plane then lo plane

    // ---- async copy of packed W into LDS (ASYNCcnt-tracked) ----
    {
        const char* gbase = (const char*)wpack;
        // 2*WFRAGS fragments * 32B = 4096 x 16B chunks; 16 uniform iterations
        for (int i = threadIdx.x; i < 4096; i += 256) {
            unsigned    ldsOff = (unsigned)(size_t)((const char*)sW) + (unsigned)(i * 16);
            const void* gp     = gbase + (size_t)i * 16;
            asm volatile("global_load_async_to_lds_b128 %0, %1, off"
                         :: "v"(ldsOff), "v"(gp) : "memory");
        }
        asm volatile("s_wait_asynccnt 0" ::: "memory");
    }
    __syncthreads();

    const int lane    = threadIdx.x & 31;
    const int wave    = threadIdx.x >> 5;
    const int rowBase = (blockIdx.x * 8 + wave) * 16;
    const int half    = lane >> 4;     // 0: K {0..7,16..23}, 1: K {8..15,24..31}
    const int lidx    = lane & 15;     // A row / B col within tile
    const int koff    = half * 8;

    // preload + split A fragments for all 4 K-tiles of this row panel
    const float* xrow = x + (size_t)(rowBase + lidx) * CIN;
    v16bf a_hi[4], a_lo[4];
    #pragma unroll
    for (int kt = 0; kt < 4; ++kt)
        cvt_split16(xrow, kt * 32 + koff, a_hi[kt], a_lo[kt]);

    #pragma unroll
    for (int nt = 0; nt < 8; ++nt) {
        const int o = nt * 16 + lidx;              // output channel
        v8f acc = {};
        #pragma unroll
        for (int kt = 0; kt < 4; ++kt) {
            const int fidx = (nt * 4 + kt) * 32 + lane;
            v16bf b_hi = sW[fidx];
            v16bf b_lo = sW[WFRAGS + fidx];
            // fp32-accurate product: lo*hi + hi*lo + hi*hi (fp32 accum)
            acc = __builtin_amdgcn_wmma_f32_16x16x32_bf16(
                      false, a_lo[kt], false, b_hi, (short)0, acc, false, false);
            acc = __builtin_amdgcn_wmma_f32_16x16x32_bf16(
                      false, a_hi[kt], false, b_lo, (short)0, acc, false, false);
            acc = __builtin_amdgcn_wmma_f32_16x16x32_bf16(
                      false, a_hi[kt], false, b_hi, (short)0, acc, false, false);
        }
        // D layout: VGPR r, lane L -> M = r + 8*(L/16), N = L%16
        #pragma unroll
        for (int r = 0; r < 8; ++r)
            xp[(size_t)(rowBase + r + half * 8) * COUT + o] = acc[r];
    }
}

// ---------------- kernel 2: per-node attention coefficients ----------------
__global__ void gat_attn_coef(const float* __restrict__ xp,
                              const float* __restrict__ att_src,
                              const float* __restrict__ att_dst,
                              float* __restrict__ a_src,
                              float* __restrict__ a_dst) {
    int t = blockIdx.x * blockDim.x + threadIdx.x;   // BN*H threads
    if (t >= BN * H) return;
    int n = t / H, h = t - n * H;
    const float* v  = xp + (size_t)n * COUT + h * CH;
    const float* as = att_src + h * CH;
    const float* ad = att_dst + h * CH;
    float s1 = 0.0f, s2 = 0.0f;
    #pragma unroll 8
    for (int c = 0; c < CH; ++c) { float f = v[c]; s1 += f * as[c]; s2 += f * ad[c]; }
    a_src[t] = s1;
    a_dst[t] = s2;
}

// ---------------- init kernels ----------------
__global__ void gat_init_state(float* __restrict__ m, float* __restrict__ denom) {
    int t = blockIdx.x * blockDim.x + threadIdx.x;   // BN*H
    if (t >= BN * H) return;
    m[t]     = -INFINITY;
    denom[t] = 0.0f;
}

__global__ void gat_init_out(float* __restrict__ out, const float* __restrict__ bias) {
    int t = blockIdx.x * blockDim.x + threadIdx.x;   // BN*COUT
    if (t >= BN * COUT) return;
    out[t] = bias[t & (COUT - 1)];
}

// ---------------- kernel 3: segment max over incoming edges ----------------
__global__ void gat_edge_max(const int* __restrict__ ei,
                             const float* __restrict__ a_src,
                             const float* __restrict__ a_dst,
                             float* __restrict__ m) {
    int t = blockIdx.x * blockDim.x + threadIdx.x;   // ETOT*H
    if (t >= ETOT * H) return;
    int e = t >> 2, h = t & 3;
    int s, d; edge_nodes(e, ei, s, d);
    float logit = lrelu(a_src[s * H + h] + a_dst[d * H + h]);
    atomicMaxFloat(&m[d * H + h], logit);
}

// ---------------- kernel 4: softmax denominator ----------------
__global__ void gat_edge_denom(const int* __restrict__ ei,
                               const float* __restrict__ a_src,
                               const float* __restrict__ a_dst,
                               const float* __restrict__ m,
                               float* __restrict__ denom) {
    int t = blockIdx.x * blockDim.x + threadIdx.x;   // ETOT*H
    if (t >= ETOT * H) return;
    int e = t >> 2, h = t & 3;
    int s, d; edge_nodes(e, ei, s, d);
    float logit = lrelu(a_src[s * H + h] + a_dst[d * H + h]);
    unsafeAtomicAdd(&denom[d * H + h], __expf(logit - m[d * H + h]));
}

// ---------------- kernel 5: weighted scatter (dominant pass) ----------------
// one wave per edge; lane handles 4 contiguous channels (head = lane/8)
__global__ __launch_bounds__(256)
void gat_edge_scatter(const int* __restrict__ ei,
                      const float* __restrict__ xp,
                      const float* __restrict__ a_src,
                      const float* __restrict__ a_dst,
                      const float* __restrict__ m,
                      const float* __restrict__ denom,
                      float* __restrict__ out) {
    int gid = blockIdx.x * blockDim.x + threadIdx.x; // ETOT*32
    int e = gid >> 5;
    if (e >= ETOT) return;
    int lane = gid & 31;
    int h    = lane >> 3;

    int s, d; edge_nodes(e, ei, s, d);
    float logit = lrelu(a_src[s * H + h] + a_dst[d * H + h]);
    float alpha = __expf(logit - m[d * H + h]) / denom[d * H + h];

    const float4 xv = *(const float4*)(xp + (size_t)s * COUT + lane * 4);
    float* op = out + (size_t)d * COUT + lane * 4;
    unsafeAtomicAdd(op + 0, xv.x * alpha);
    unsafeAtomicAdd(op + 1, xv.y * alpha);
    unsafeAtomicAdd(op + 2, xv.z * alpha);
    unsafeAtomicAdd(op + 3, xv.w * alpha);
}

// ---------------- host-side launcher ----------------
extern "C" void kernel_launch(void* const* d_in, const int* in_sizes, int n_in,
                              void* d_out, int out_size, void* d_ws, size_t ws_size,
                              hipStream_t stream) {
    const float* x        = (const float*)d_in[0];   // [B,N,CIN]
    const int*   ei       = (const int*)  d_in[1];   // [2,E]
    const float* W        = (const float*)d_in[2];   // [COUT,CIN]
    const float* att_src  = (const float*)d_in[3];   // [1,H,CH]
    const float* att_dst  = (const float*)d_in[4];   // [1,H,CH]
    const float* bias     = (const float*)d_in[5];   // [COUT]
    float*       out      = (float*)d_out;           // [B,N,COUT]

    // workspace layout
    float* xp     = (float*)d_ws;                    // BN*COUT floats
    float* a_src  = xp    + (size_t)BN * COUT;       // BN*H
    float* a_dst  = a_src + (size_t)BN * H;
    float* mbuf   = a_dst + (size_t)BN * H;
    float* denom  = mbuf  + (size_t)BN * H;
    v16bf* wpack  = (v16bf*)(denom + (size_t)BN * H);// 2*WFRAGS fragments (64KB),
                                                     // byte offset 18,432,000 (32B aligned)

    // 0) pack W into WMMA fragment layout (bf16 hi/lo)
    gat_pack_w<<<(WFRAGS + 255) / 256, 256, 0, stream>>>(W, wpack);

    // 1) projection GEMM (WMMA): 250 blocks x 8 waves x 16 rows = 32000 rows
    gat_gemm_wmma<<<BN / (8 * 16), 256, 0, stream>>>(x, wpack, xp);

    // 2) per-node attention coefficients
    gat_attn_coef<<<(BN * H + 255) / 256, 256, 0, stream>>>(xp, att_src, att_dst,
                                                            a_src, a_dst);

    // 3) init softmax state + output (bias broadcast)
    gat_init_state<<<(BN * H + 255) / 256, 256, 0, stream>>>(mbuf, denom);
    gat_init_out<<<(BN * COUT + 255) / 256, 256, 0, stream>>>(out, bias);

    // 4) segment max
    gat_edge_max<<<(ETOT * H + 255) / 256, 256, 0, stream>>>(ei, a_src, a_dst, mbuf);

    // 5) softmax denominator
    gat_edge_denom<<<(ETOT * H + 255) / 256, 256, 0, stream>>>(ei, a_src, a_dst,
                                                               mbuf, denom);

    // 6) weighted aggregation into out
    gat_edge_scatter<<<((size_t)ETOT * 32 + 255) / 256, 256, 0, stream>>>(
        ei, xp, a_src, a_dst, mbuf, denom, out);
}
